// DecoderDynamicTokenSideEmbedding_55602646614060
// MI455X (gfx1250) — matrix operand order repaired and checked by reading, not compiled, not run on hardware
//
#include <hip/hip_runtime.h>
#include <hip/hip_fp16.h>

#define BB 16
#define LL 8192
#define NTOK (BB * LL)       // 131072
#define DENOM 8191.0f
#define NTILES (NTOK / 16)   // 8192
#define NT8 (NTILES / 8)     // 1024 block iterations

typedef __attribute__((ext_vector_type(16))) _Float16 v16h;
typedef __attribute__((ext_vector_type(8)))  float    v8f;

union Frag16 {
    v16h h;
    uint4 u[2];
    unsigned u32[8];
};

// ---------------------------------------------------------------------------
// Kernel 1: exclusive prefix scans (global / per-family / per-group)
// One block per batch row. 64 threads: cooperative LDS staging, then 49
// channel threads walk the chunk serially, each owning one running pair.
// ---------------------------------------------------------------------------
__global__ __launch_bounds__(64) void scan_kernel(
    const int* __restrict__ token_ids, const int* __restrict__ var_ids,
    const float* __restrict__ t_has_int, const float* __restrict__ t_log,
    const int* __restrict__ v_fam, const int* __restrict__ v_grp,
    float* __restrict__ S)
{
    __shared__ float s_vf[2048];
    __shared__ float s_lv[2048];
    __shared__ int   s_id[2048];
    const int b = blockIdx.x;
    const int tid = threadIdx.x;
    const size_t rowbase = (size_t)b * LL;
    float c = 0.0f, s = 0.0f;

    for (int chunk = 0; chunk < LL; chunk += 2048) {
        for (int i = tid; i < 2048; i += 64) {
            int t = chunk + i;
            int tok = token_ids[rowbase + t];
            int var = var_ids[rowbase + t];
            float hi = t_has_int[tok];
            bool valid = (hi > 0.0f) && (tok != 0) && (tok != 1) && (tok != 2);
            float vf = valid ? 1.0f : 0.0f;
            s_vf[i] = vf;
            s_lv[i] = t_log[tok] * vf;
            s_id[i] = (v_fam[var] & 0xff) | (v_grp[var] << 8);
        }
        __syncthreads();

        float* Sc = S + (rowbase + chunk) * 6;
        if (tid == 0) {                       // global count / logsum
            for (int i = 0; i < 2048; ++i) {
                float* p = Sc + (size_t)i * 6;
                p[0] = c; p[1] = s;
                c += s_vf[i]; s += s_lv[i];
            }
        } else if (tid <= 16) {               // family tid-1
            int f = tid - 1;
            for (int i = 0; i < 2048; ++i) {
                if ((s_id[i] & 0xff) == f) {
                    float* p = Sc + (size_t)i * 6;
                    p[2] = c; p[3] = s;
                    c += s_vf[i]; s += s_lv[i];
                }
            }
        } else if (tid == 17) {               // group 0 => psg forced to 0
            for (int i = 0; i < 2048; ++i) {
                if ((s_id[i] >> 8) == 0) {
                    float* p = Sc + (size_t)i * 6;
                    p[4] = 0.0f; p[5] = 0.0f;
                }
            }
        } else if (tid <= 48) {               // groups 1..31
            int g = tid - 17;
            for (int i = 0; i < 2048; ++i) {
                if ((s_id[i] >> 8) == g) {
                    float* p = Sc + (size_t)i * 6;
                    p[4] = c; p[5] = s;
                    c += s_vf[i]; s += s_lv[i];
                }
            }
        }
        __syncthreads();
    }
}

// ---------------------------------------------------------------------------
// Kernel 2: assemble 37 features + LayerNorm -> f16 X[tokens][64] (zero pad)
// ---------------------------------------------------------------------------
__global__ __launch_bounds__(256) void feat_kernel(
    const int* __restrict__ token_ids, const int* __restrict__ var_ids,
    const float* __restrict__ t_has_int, const float* __restrict__ t_signed,
    const float* __restrict__ t_log, const float* __restrict__ t_zero,
    const float* __restrict__ t_one, const float* __restrict__ t_pow2,
    const float* __restrict__ v_outer, const float* __restrict__ v_inner,
    const float* __restrict__ v_has_outer, const float* __restrict__ v_has_inner,
    const int* __restrict__ v_fam,
    const float* __restrict__ ln_g, const float* __restrict__ ln_b,
    const float* __restrict__ S, __half* __restrict__ X)
{
    int id = blockIdx.x * blockDim.x + threadIdx.x;
    if (id >= NTOK) return;
    int t = id & (LL - 1);
    int tok = token_ids[id];
    int var = var_ids[id];

    float feat[37];
    feat[0] = t_has_int[tok];
    feat[1] = t_signed[tok];
    feat[2] = t_log[tok];
    feat[3] = t_zero[tok];
    feat[4] = t_one[tok];
    feat[5] = t_pow2[tok];
    int fam = v_fam[var];
#pragma unroll
    for (int j = 0; j < 16; ++j) feat[6 + j] = (fam == j) ? 1.0f : 0.0f;
    feat[22] = v_outer[var];
    feat[23] = v_inner[var];
    feat[24] = v_has_outer[var];
    feat[25] = v_has_inner[var];

    const float inv = 1.0f / DENOM;
    size_t so = (size_t)id * 6;
    float pc = S[so + 0], pl = S[so + 1];
    float fc = S[so + 2], fl = S[so + 3];
    float gc = S[so + 4], gl = S[so + 5];
    float cnt_floor = fmaxf(pc, 1.0f);
    feat[26] = (float)t * inv;
    feat[27] = pc * inv;
    feat[28] = pl * inv;
    feat[29] = fc * inv;
    feat[30] = fl * inv;
    feat[31] = fc / cnt_floor;
    feat[32] = gc * inv;
    feat[33] = gl * inv;
    feat[34] = gc / cnt_floor;
    float ptl = 0.0f, pts = 0.0f;
    if (t > 0) {
        int pt = token_ids[id - 1];
        ptl = t_log[pt];
        pts = t_signed[pt];
    }
    feat[35] = ptl;
    feat[36] = pts;

    float mu = 0.0f;
#pragma unroll
    for (int i = 0; i < 37; ++i) mu += feat[i];
    mu *= (1.0f / 37.0f);
    float var2 = 0.0f;
#pragma unroll
    for (int i = 0; i < 37; ++i) { float d = feat[i] - mu; var2 += d * d; }
    var2 *= (1.0f / 37.0f);
    float rs = rsqrtf(var2 + 1e-5f);

    __half2* row = (__half2*)(X + (size_t)id * 64);
#pragma unroll
    for (int i = 0; i < 32; ++i) {
        int i0 = 2 * i, i1 = 2 * i + 1;
        float x0 = (i0 < 37) ? ((feat[i0] - mu) * rs * ln_g[i0] + ln_b[i0]) : 0.0f;
        float x1 = (i1 < 37) ? ((feat[i1] - mu) * rs * ln_g[i1] + ln_b[i1]) : 0.0f;
        row[i] = __halves2half2(__float2half_rn(x0), __float2half_rn(x1));
    }
}

// ---------------------------------------------------------------------------
// Kernel 3: fused MLP via WMMA f16, cooperative GEMM1.
// 8 waves/block. Per block iteration (8 token tiles):
//   - wave w runs GEMM1+GELU for token tile (tile8*8 + w) into Hstage[w]
//   - barrier
//   - every wave runs GEMM2 over all 8 H tiles for its own 6 N-tiles of D=768,
//     W2 fragments register-resident for the whole kernel.
// All global/LDS accesses use one per-lane base + compile-time offsets.
// ---------------------------------------------------------------------------
__global__ __launch_bounds__(256) void mlp_kernel(
    const __half* __restrict__ X, const float* __restrict__ W1,
    const float* __restrict__ b1, const float* __restrict__ W2,
    const float* __restrict__ b2, const float* __restrict__ scale_p,
    float* __restrict__ out)
{
    __shared__ __align__(16) _Float16 W1c[64 * 68];      // [n][k], stride 68
    __shared__ __align__(16) _Float16 Hst[8 * 16 * 72];  // per-tile-slot [16][72]
    const int tid = threadIdx.x;
    const int lane = tid & 31;
    const int w = tid >> 5;
    const int hsel = (lane & 16) ? 1 : 0;   // upper half-wave
    const int lm = lane & 15;
    const float scale = scale_p[0];

    // Stage W1 into LDS as [n][k] with K zero-padded 37..63.
    for (int idx = tid; idx < 64 * 64; idx += 256) {
        int n = idx >> 6, k = idx & 63;
        float v = (k < 37) ? W1[k * 64 + n] : 0.0f;
        W1c[n * 68 + k] = (_Float16)v;
    }

    // Preload W2 B-fragments (32x16 f16: lanes0-15 K+0..7,16..23, lanes16-31 K+8..15,24..31).
    Frag16 w2f[6][2];
    float b2s[6];
#pragma unroll
    for (int j = 0; j < 6; ++j) {
        int n = 16 * (6 * w + j) + lm;
        b2s[j] = scale * b2[n];
#pragma unroll
        for (int kf = 0; kf < 2; ++kf) {
            int kbase = 32 * kf + (hsel ? 8 : 0);
#pragma unroll
            for (int v = 0; v < 8; ++v) {
                int k = kbase + 2 * v + ((v >= 4) ? 8 : 0);
                w2f[j][kf].h[2 * v]     = (_Float16)W2[(size_t)k * 768 + n];
                w2f[j][kf].h[2 * v + 1] = (_Float16)W2[(size_t)(k + 1) * 768 + n];
            }
        }
    }
    float b1v[4];
#pragma unroll
    for (int j = 0; j < 4; ++j) b1v[j] = b1[16 * j + lm];
    __syncthreads();

    // Per-lane invariant offsets (all compile into the loop bases once).
    const int kh = hsel ? 8 : 0;                       // A-layout half-offset (halves)
    const int w1_base = lm * 68 + hsel * 8;            // W1c read base (halves)
    const int hw_base = hsel * 8 * 72 + lm;            // Hstage write base (halves)
    const int ha_base = lm * 72 + kh;                  // Hstage A-read base (halves)
    const size_t out_lane = (size_t)hsel * 8 * 768 + 96 * w + lm;  // floats
    const size_t x_lane = (size_t)lm * 64 + kh;        // halves

    for (int t8 = blockIdx.x; t8 < NT8; t8 += gridDim.x) {
        const int mytile = t8 * 8 + w;

        // ---- GEMM1 (X @ W1) + bias + exact GELU for this wave's token tile ----
        const __half* px = X + (size_t)mytile * 16 * 64 + x_lane;
        Frag16 ax[2];
        ax[0].u[0] = *(const uint4*)(px);
        ax[0].u[1] = *(const uint4*)(px + 16);
        ax[1].u[0] = *(const uint4*)(px + 32);
        ax[1].u[1] = *(const uint4*)(px + 48);

        _Float16* hw = &Hst[w * 16 * 72];
#pragma unroll
        for (int j = 0; j < 4; ++j) {
            v8f acc = {};
#pragma unroll
            for (int kf = 0; kf < 2; ++kf) {
                const unsigned* w1u = (const unsigned*)&W1c[w1_base + j * 1088 + kf * 32];
                Frag16 bf;
#pragma unroll
                for (int v = 0; v < 8; ++v) bf.u32[v] = w1u[v + (v & 4)];
                acc = __builtin_amdgcn_wmma_f32_16x16x32_f16(
                    false, ax[kf].h, false, bf.h, (short)0, acc, false, false);
            }
#pragma unroll
            for (int r = 0; r < 8; ++r) {
                float hv = acc[r] + b1v[j];
                float g = 0.5f * hv * (1.0f + erff(hv * 0.70710678118654752f));
                hw[hw_base + r * 72 + 16 * j] = (_Float16)g;
            }
        }
        __syncthreads();

        // ---- GEMM2 (H @ W2) over all 8 staged tiles, this wave's 6 N-tiles ----
        float* po = out + (size_t)t8 * 8 * 16 * 768 + out_lane;
        for (int tt = 0; tt < 8; ++tt) {
            const _Float16* hr = &Hst[tt * 16 * 72 + ha_base];
            Frag16 ah[2];
            ah[0].u[0] = *(const uint4*)(hr);
            ah[0].u[1] = *(const uint4*)(hr + 16);
            ah[1].u[0] = *(const uint4*)(hr + 32);
            ah[1].u[1] = *(const uint4*)(hr + 48);
            float* pt = po + (size_t)tt * 16 * 768;
#pragma unroll
            for (int j = 0; j < 6; ++j) {
                v8f c = {};
                c = __builtin_amdgcn_wmma_f32_16x16x32_f16(
                    false, ah[0].h, false, w2f[j][0].h, (short)0, c, false, false);
                c = __builtin_amdgcn_wmma_f32_16x16x32_f16(
                    false, ah[1].h, false, w2f[j][1].h, (short)0, c, false, false);
#pragma unroll
                for (int r = 0; r < 8; ++r) {
                    pt[r * 768 + 16 * j] = fmaf(scale, c[r], b2s[j]);
                }
            }
        }
        __syncthreads();
    }
}

// ---------------------------------------------------------------------------
extern "C" void kernel_launch(void* const* d_in, const int* in_sizes, int n_in,
                              void* d_out, int out_size, void* d_ws, size_t ws_size,
                              hipStream_t stream) {
    (void)in_sizes; (void)n_in; (void)out_size; (void)ws_size;
    const int*   token_ids  = (const int*)d_in[0];
    const int*   var_ids    = (const int*)d_in[1];
    const float* t_has_int  = (const float*)d_in[2];
    const float* t_signed   = (const float*)d_in[3];
    const float* t_log      = (const float*)d_in[4];
    const float* t_zero     = (const float*)d_in[5];
    const float* t_one      = (const float*)d_in[6];
    const float* t_pow2     = (const float*)d_in[7];
    // d_in[8] var_family_onehot: recomputed from var_family_id instead
    const float* v_outer    = (const float*)d_in[9];
    const float* v_inner    = (const float*)d_in[10];
    const float* v_has_out  = (const float*)d_in[11];
    const float* v_has_inn  = (const float*)d_in[12];
    const int*   v_fam      = (const int*)d_in[13];
    const int*   v_grp      = (const int*)d_in[14];
    const float* ln_g       = (const float*)d_in[15];
    const float* ln_b       = (const float*)d_in[16];
    const float* W1         = (const float*)d_in[17];
    const float* b1         = (const float*)d_in[18];
    const float* W2         = (const float*)d_in[19];
    const float* b2         = (const float*)d_in[20];
    const float* scale      = (const float*)d_in[21];

    float*  S = (float*)d_ws;                                   // NTOK*6 f32 = 3.1 MB
    __half* X = (__half*)((char*)d_ws + (size_t)NTOK * 6 * 4);  // NTOK*64 f16 = 16.8 MB

    scan_kernel<<<BB, 64, 0, stream>>>(token_ids, var_ids, t_has_int, t_log,
                                       v_fam, v_grp, S);
    feat_kernel<<<NTOK / 256, 256, 0, stream>>>(
        token_ids, var_ids, t_has_int, t_signed, t_log, t_zero, t_one, t_pow2,
        v_outer, v_inner, v_has_out, v_has_inn, v_fam, ln_g, ln_b, S, X);
    mlp_kernel<<<1024, 256, 0, stream>>>(X, W1, b1, W2, b2, scale, (float*)d_out);
}
